// TemporalMambaFusionNet_89945205113393
// MI455X (gfx1250) — matrix operand order repaired and checked by reading, not compiled, not run on hardware
//
#include <hip/hip_runtime.h>
#include <hip/hip_bf16.h>
#include <math.h>

// ---------------------------------------------------------------------------
// TemporalMambaFusionNet on MI455X (gfx1250, wave32, WMMA + TDM).
// All conv / linear layers -> implicit GEMM on v_wmma_f32_16x16x32_f16.
// LDS tiles double-buffered and staged by the Tensor Data Mover with partial
// s_wait_tensorcnt waits so tile k+1's DMA overlaps tile k's WMMA.
// GroupNorm+GELU (+residual) fused in one pass; activations carried as f16.
// ---------------------------------------------------------------------------

typedef _Float16 f16;
typedef __attribute__((ext_vector_type(16))) _Float16 v16h;
typedef __attribute__((ext_vector_type(8)))  _Float16 v8h;
typedef __attribute__((ext_vector_type(8)))  float    v8f;
typedef __attribute__((ext_vector_type(4)))  unsigned int v4u;
typedef __attribute__((ext_vector_type(8)))  int      v8i;
typedef __attribute__((ext_vector_type(4)))  int      v4i;

__device__ __forceinline__ float gelu_f(float x) {
    return 0.5f * x * (1.0f + erff(x * 0.70710678118654752f));
}
__device__ __forceinline__ float sigmoid_f(float x) {
    return 1.0f / (1.0f + __expf(-x));
}

// ---------------------------------------------------------------------------
// Tensor Data Mover: 2D tile load (global, row-major, elem=2B) -> LDS.
// D# built per CDNA5 ISA ch.8: group0 = {count|flags, lds_addr, global_addr,
// type=2}; group1 = {data_size, tensor dims (remaining extent, HW zero-fills
// OOB), tile dims, row stride}.  Issued by one wave; TENSORcnt + barrier sync.
// ---------------------------------------------------------------------------
#if __has_builtin(__builtin_amdgcn_tensor_load_to_lds) && __has_builtin(__builtin_amdgcn_s_wait_tensorcnt)
#define HAVE_TDM 1
#endif

#ifdef HAVE_TDM
__device__ __forceinline__ void tdm_load_2d_f16(
    unsigned lds_off, const void* gaddr,
    unsigned rem_elems, unsigned rem_rows,
    unsigned tile_elems, unsigned tile_rows,
    unsigned row_stride_elems)
{
    unsigned long long ga = (unsigned long long)(size_t)gaddr;
    v4u g0;
    g0.x = 1u;                                        // count=1 (valid user D#)
    g0.y = lds_off;                                   // LDS byte address
    g0.z = (unsigned)(ga & 0xFFFFFFFFu);              // global_addr[31:0]
    g0.w = (unsigned)((ga >> 32) & 0x01FFFFFFu)       // global_addr[56:32]
         | 0x80000000u;                               // type=2 ("image")
    v8i g1;
    g1[0] = (int)(1u << 16);                          // data_size=1 -> 2 bytes
    g1[1] = (int)((rem_elems & 0xFFFFu) << 16);       // tensor_dim0[15:0]
    g1[2] = (int)(((rem_elems >> 16) & 0xFFFFu)       // tensor_dim0[31:16]
         | ((rem_rows & 0xFFFFu) << 16));             // tensor_dim1[15:0]
    g1[3] = (int)(((rem_rows >> 16) & 0xFFFFu)        // tensor_dim1[31:16]
         | ((tile_elems & 0xFFFFu) << 16));           // tile_dim0
    g1[4] = (int)(tile_rows & 0xFFFFu);               // tile_dim1 (tile_dim2=0)
    g1[5] = (int)row_stride_elems;                    // tensor_dim0_stride[31:0]
    g1[6] = 0;                                        // stride hi / dim1_stride lo
    g1[7] = 0;
    v4i z4 = {0, 0, 0, 0};
#if __has_include(<hip/amd_detail/amd_gfx1250_TDM.h>)
    v8i z8 = {0, 0, 0, 0, 0, 0, 0, 0};
    __builtin_amdgcn_tensor_load_to_lds(g0, g1, z4, z4, z8, 0);   // 6-arg toolchain
#else
    __builtin_amdgcn_tensor_load_to_lds(g0, g1, z4, z4, 0);       // 5-arg toolchain
#endif
}
#endif

// ---------------------------------------------------------------------------
// Implicit-GEMM conv:  out[n,m] = sum_k  im2col(in)[m,k] * wt[n,k]
//   in : f16 [img][Cin][Hin][Win]     wt : f16 [Cout][Cin*KSZ*KSZ]
//   out: f32 [img][Cout][Hout][Wout]
// Block tile 128(M=pixels) x 64(N=cout), K-step 32, 8 waves x 4 WMMA / step.
// B tiles double-buffered via TDM (wave 0, s_wait_tensorcnt(1) pipelining);
// A tile im2col-gathered by all waves with incremental (c,ky,kx) decomposition.
// ---------------------------------------------------------------------------
template <int KSZ>
__global__ __launch_bounds__(256) void k_conv_gemm(
    const f16* __restrict__ in, const f16* __restrict__ wt, float* __restrict__ out,
    int Cin, int Hin, int Win_, int Cout, int Hout, int Wout, int stride)
{
    constexpr int KHW = KSZ * KSZ;
    constexpr int PAD = KSZ / 2;
    __shared__ alignas(16) f16 As[128 * 32];
    __shared__ alignas(16) f16 Bs[2][64 * 32];
    const int tid  = threadIdx.x;
    const int wv   = tid >> 5;
    const int lane = tid & 31;
    const int tile_m = blockIdx.x * 128;
    const int tile_n = blockIdx.y * 64;
    const int HWo = Hout * Wout;
    const int K   = Cin * KHW;
    const int HinWin = Hin * Win_;
    const f16* inI = in + (size_t)blockIdx.z * Cin * HinWin;
    const int nsteps = (K + 31) / 32;

    v8f a0 = {}, a1 = {}, a2 = {}, a3 = {};

    const int arow = tid >> 1;
    const int acol = (tid & 1) * 16;
    const int am   = tile_m + arow;
    const int aoy  = am / Wout;
    const int aox  = am - aoy * Wout;
    const int ybase = aoy * stride - PAD;
    const int xbase = aox * stride - PAD;
    const bool mvalid = (am < HWo);
#ifndef HAVE_TDM
    const int brow = tid >> 2;
    const int bcol = (tid & 3) * 8;
    const int bn   = tile_n + brow;
#endif

#ifdef HAVE_TDM
    if (wv == 0) {     // prologue: B tile 0 in flight
        tdm_load_2d_f16((unsigned)(size_t)&Bs[0][0], wt + (size_t)tile_n * K,
                        (unsigned)K, (unsigned)(Cout - tile_n), 32u, 64u, (unsigned)K);
    }
#endif
    for (int step = 0; step < nsteps; ++step) {
        const int kb  = step * 32;
        const int cur = step & 1;
        // ---- stage A (im2col gather, incremental k decomposition) ----
        {
            int k0 = kb + acol;
            int c  = k0 / KHW;            // constant divisor -> mul/shift
            int r  = k0 - c * KHW;
            int ky = r / KSZ;
            int kx = r - ky * KSZ;
            #pragma unroll
            for (int i = 0; i < 16; ++i) {
                f16 v = (f16)0.f;
                if (mvalid && (k0 + i) < K) {
                    int iy = ybase + ky;
                    int ix = xbase + kx;
                    if (iy >= 0 && iy < Hin && ix >= 0 && ix < Win_)
                        v = inI[(size_t)c * HinWin + (size_t)iy * Win_ + ix];
                }
                As[arow * 32 + acol + i] = v;
                if (++kx == KSZ) { kx = 0; if (++ky == KSZ) { ky = 0; ++c; } }
            }
        }
#ifdef HAVE_TDM
        // ---- pipeline next B tile; wait only for the tile we consume ----
        if (wv == 0) {
            if (step + 1 < nsteps) {
                int kn = kb + 32;
                tdm_load_2d_f16((unsigned)(size_t)&Bs[cur ^ 1][0],
                                wt + (size_t)tile_n * K + kn,
                                (unsigned)(K - kn), (unsigned)(Cout - tile_n),
                                32u, 64u, (unsigned)K);
                __builtin_amdgcn_s_wait_tensorcnt((short)1);
            } else {
                __builtin_amdgcn_s_wait_tensorcnt((short)0);
            }
        }
#else
        if (bn < Cout && kb + 32 < K)
            __builtin_prefetch(&wt[(size_t)bn * K + kb + 32], 0, 1);
        #pragma unroll
        for (int i = 0; i < 8; ++i) {
            int k = kb + bcol + i;
            Bs[0][brow * 32 + bcol + i] = (bn < Cout && k < K) ? wt[(size_t)bn * K + k] : (f16)0.f;
        }
        (void)cur;
#endif
        __syncthreads();

        // ---- fragments (ISA VGPR layouts, 16B-aligned LDS reads) ----
        v16h af;
        {
            const f16* pa = &As[(16 * wv + (lane & 15)) * 32 + ((lane & 16) ? 8 : 0)];
            v8h lo2 = *(const v8h*)pa;
            v8h hi2 = *(const v8h*)(pa + 16);
            #pragma unroll
            for (int i = 0; i < 8; ++i) { af[i] = lo2[i]; af[8 + i] = hi2[i]; }
        }
#ifdef HAVE_TDM
        const f16* bbase = &Bs[cur][0];
#else
        const f16* bbase = &Bs[0][0];
#endif
        #pragma unroll
        for (int s = 0; s < 4; ++s) {
            const f16* pb = bbase + (s * 16 + (lane & 15)) * 32 + ((lane & 16) ? 8 : 0);
            v8h lo2 = *(const v8h*)pb;
            v8h hi2 = *(const v8h*)(pb + 16);
            v16h bf;
            #pragma unroll
            for (int i = 0; i < 8; ++i) { bf[i] = lo2[i]; bf[8 + i] = hi2[i]; }
            if (s == 0)      a0 = __builtin_amdgcn_wmma_f32_16x16x32_f16(false, af, false, bf, (short)0, a0, false, false);
            else if (s == 1) a1 = __builtin_amdgcn_wmma_f32_16x16x32_f16(false, af, false, bf, (short)0, a1, false, false);
            else if (s == 2) a2 = __builtin_amdgcn_wmma_f32_16x16x32_f16(false, af, false, bf, (short)0, a2, false, false);
            else             a3 = __builtin_amdgcn_wmma_f32_16x16x32_f16(false, af, false, bf, (short)0, a3, false, false);
        }
        __syncthreads();
    }

    // ---- epilogue: NCHW scatter ----
    float* outI = out + (size_t)blockIdx.z * Cout * HWo;
    const int madd = (lane & 16) ? 8 : 0;
    const int n0   = tile_n + (lane & 15);
    #pragma unroll
    for (int r = 0; r < 8; ++r) {
        int m = tile_m + 16 * wv + r + madd;
        if (m >= HWo) continue;
        if (n0      < Cout) outI[(size_t)(n0     ) * HWo + m] = a0[r];
        if (n0 + 16 < Cout) outI[(size_t)(n0 + 16) * HWo + m] = a1[r];
        if (n0 + 32 < Cout) outI[(size_t)(n0 + 32) * HWo + m] = a2[r];
        if (n0 + 48 < Cout) outI[(size_t)(n0 + 48) * HWo + m] = a3[r];
    }
}

// ---------------------------------------------------------------------------
// Token GEMM: out[m,n] = act( A[m,:] . W[n,:] + bias[n] + resid[m,n] )
//   A: f16 [M][K] row-major, W: f16 [N][K] row-major, out: f32 [M][N].
// Both A and B tiles are plain 2D tiles -> double-buffered TDM pipeline:
// issue tile k+1, s_wait_tensorcnt(2) covers only tile k.
// act: 0=none 1=gelu 2=sigmoid
// ---------------------------------------------------------------------------
__global__ __launch_bounds__(256) void k_lin_gemm(
    const f16* __restrict__ A, const f16* __restrict__ W,
    const float* __restrict__ bias, const float* __restrict__ resid,
    float* __restrict__ out, int M, int K, int N, int act)
{
    __shared__ alignas(16) f16 As[2][128 * 32];
    __shared__ alignas(16) f16 Bs[2][64 * 32];
    const int tid = threadIdx.x, wv = tid >> 5, lane = tid & 31;
    const int tile_m = blockIdx.x * 128, tile_n = blockIdx.y * 64;
    const int nsteps = (K + 31) / 32;
    v8f a0 = {}, a1 = {}, a2 = {}, a3 = {};
#ifndef HAVE_TDM
    const int arow = tid >> 1, acol = (tid & 1) * 16, am = tile_m + arow;
    const int brow = tid >> 2, bcol = (tid & 3) * 8, bn = tile_n + brow;
#endif

#ifdef HAVE_TDM
    if (wv == 0) {     // prologue: tile 0 (A+B) in flight
        tdm_load_2d_f16((unsigned)(size_t)&As[0][0], A + (size_t)tile_m * K,
                        (unsigned)K, (unsigned)(M - tile_m), 32u, 128u, (unsigned)K);
        tdm_load_2d_f16((unsigned)(size_t)&Bs[0][0], W + (size_t)tile_n * K,
                        (unsigned)K, (unsigned)(N - tile_n), 32u, 64u, (unsigned)K);
    }
#endif
    for (int step = 0; step < nsteps; ++step) {
        const int cur = step & 1;
#ifdef HAVE_TDM
        if (wv == 0) {
            if (step + 1 < nsteps) {
                int kn = (step + 1) * 32;
                tdm_load_2d_f16((unsigned)(size_t)&As[cur ^ 1][0],
                                A + (size_t)tile_m * K + kn,
                                (unsigned)(K - kn), (unsigned)(M - tile_m),
                                32u, 128u, (unsigned)K);
                tdm_load_2d_f16((unsigned)(size_t)&Bs[cur ^ 1][0],
                                W + (size_t)tile_n * K + kn,
                                (unsigned)(K - kn), (unsigned)(N - tile_n),
                                32u, 64u, (unsigned)K);
                __builtin_amdgcn_s_wait_tensorcnt((short)2);   // tile k done, k+1 in flight
            } else {
                __builtin_amdgcn_s_wait_tensorcnt((short)0);
            }
        }
#else
        const int kb = step * 32;
        #pragma unroll
        for (int i = 0; i < 16; ++i) {
            int k = kb + acol + i;
            As[0][arow * 32 + acol + i] = (am < M && k < K) ? A[(size_t)am * K + k] : (f16)0.f;
        }
        if (bn < N && kb + 32 < K)
            __builtin_prefetch(&W[(size_t)bn * K + kb + 32], 0, 1);
        #pragma unroll
        for (int i = 0; i < 8; ++i) {
            int k = kb + bcol + i;
            Bs[0][brow * 32 + bcol + i] = (bn < N && k < K) ? W[(size_t)bn * K + k] : (f16)0.f;
        }
        (void)cur;
#endif
        __syncthreads();
#ifdef HAVE_TDM
        const f16* abase = &As[cur][0];
        const f16* bbase = &Bs[cur][0];
#else
        const f16* abase = &As[0][0];
        const f16* bbase = &Bs[0][0];
#endif
        v16h af;
        {
            const f16* pa = abase + (16 * wv + (lane & 15)) * 32 + ((lane & 16) ? 8 : 0);
            v8h lo2 = *(const v8h*)pa;
            v8h hi2 = *(const v8h*)(pa + 16);
            #pragma unroll
            for (int i = 0; i < 8; ++i) { af[i] = lo2[i]; af[8 + i] = hi2[i]; }
        }
        #pragma unroll
        for (int s = 0; s < 4; ++s) {
            const f16* pb = bbase + (s * 16 + (lane & 15)) * 32 + ((lane & 16) ? 8 : 0);
            v8h lo2 = *(const v8h*)pb;
            v8h hi2 = *(const v8h*)(pb + 16);
            v16h bf;
            #pragma unroll
            for (int i = 0; i < 8; ++i) { bf[i] = lo2[i]; bf[8 + i] = hi2[i]; }
            if (s == 0)      a0 = __builtin_amdgcn_wmma_f32_16x16x32_f16(false, af, false, bf, (short)0, a0, false, false);
            else if (s == 1) a1 = __builtin_amdgcn_wmma_f32_16x16x32_f16(false, af, false, bf, (short)0, a1, false, false);
            else if (s == 2) a2 = __builtin_amdgcn_wmma_f32_16x16x32_f16(false, af, false, bf, (short)0, a2, false, false);
            else             a3 = __builtin_amdgcn_wmma_f32_16x16x32_f16(false, af, false, bf, (short)0, a3, false, false);
        }
        __syncthreads();
    }

    const int madd = (lane & 16) ? 8 : 0;
    const int nl   = lane & 15;
    #pragma unroll
    for (int r = 0; r < 8; ++r) {
        int m = tile_m + 16 * wv + r + madd;
        if (m >= M) continue;
        #pragma unroll
        for (int s = 0; s < 4; ++s) {
            int n = tile_n + s * 16 + nl;
            if (n >= N) continue;
            float v = (s == 0) ? a0[r] : (s == 1) ? a1[r] : (s == 2) ? a2[r] : a3[r];
            if (bias)  v += bias[n];
            if (resid) v += resid[(size_t)m * N + n];
            if (act == 1) v = gelu_f(v);
            else if (act == 2) v = sigmoid_f(v);
            out[(size_t)m * N + n] = v;
        }
    }
}

// ---------------------------------------------------------------------------
// GroupNorm + GELU (+ optional residual add AFTER gelu), dual f32/f16 output.
// grid = (num_groups, num_images); x layout [img][C][HW]
// ---------------------------------------------------------------------------
__global__ __launch_bounds__(256) void k_gn_gelu(
    const float* __restrict__ x, const float* __restrict__ gamma, const float* __restrict__ beta,
    const float* __restrict__ resid, float* __restrict__ o32, f16* __restrict__ o16,
    int C, int HW, int ng)
{
    __shared__ float rs[256], rq[256];
    const int grp = blockIdx.x, img = blockIdx.y;
    const int cpg = C / ng;
    const size_t base = (size_t)img * C * HW + (size_t)grp * cpg * HW;
    const long n = (long)cpg * HW;
    float s = 0.f, q = 0.f;
    for (long i = threadIdx.x; i < n; i += 256) { float v = x[base + i]; s += v; q += v * v; }
    rs[threadIdx.x] = s; rq[threadIdx.x] = q;
    __syncthreads();
    for (int st = 128; st > 0; st >>= 1) {
        if ((int)threadIdx.x < st) { rs[threadIdx.x] += rs[threadIdx.x + st]; rq[threadIdx.x] += rq[threadIdx.x + st]; }
        __syncthreads();
    }
    const float mean = rs[0] / (float)n;
    const float var  = rq[0] / (float)n - mean * mean;
    const float inv  = rsqrtf(var + 1e-5f);
    for (long i = threadIdx.x; i < n; i += 256) {
        int c = grp * cpg + (int)(i / HW);
        float v = (x[base + i] - mean) * inv * gamma[c] + beta[c];
        v = gelu_f(v);
        if (resid) v += resid[base + i];
        if (o32) o32[base + i] = v;
        if (o16) o16[base + i] = (f16)v;
    }
}

// ---------------------------------------------------------------------------
// Pointwise helpers
// ---------------------------------------------------------------------------
__global__ void k_f32_to_f16(const float* __restrict__ a, f16* __restrict__ b, long n) {
    long i = (long)blockIdx.x * blockDim.x + threadIdx.x;
    if (i < n) b[i] = (f16)a[i];
}

__global__ void k_bias_act(float* x, const float* __restrict__ bias, int C, long HW, long total, int act) {
    long i = (long)blockIdx.x * blockDim.x + threadIdx.x;
    if (i >= total) return;
    int c = (int)((i / HW) % C);
    float v = x[i] + bias[c];
    if (act == 2) v = sigmoid_f(v);
    x[i] = v;
}

__global__ void k_conf(const float* __restrict__ in, float* __restrict__ out, long n) {
    long i = (long)blockIdx.x * blockDim.x + threadIdx.x;
    if (i < n) out[i] = fminf(fmaxf(1.0f - in[i], 0.0f), 1.0f);
}

// plane-wise half-pixel bilinear, in/out: [P][Hi][Wi] -> [P][Ho][Wo]
__global__ void k_resize(const float* __restrict__ in, float* __restrict__ out,
                         int P, int Hi, int Wi, int Ho, int Wo) {
    long idx = (long)blockIdx.x * blockDim.x + threadIdx.x;
    long total = (long)P * Ho * Wo;
    if (idx >= total) return;
    int xo = (int)(idx % Wo);
    int yo = (int)((idx / Wo) % Ho);
    int p  = (int)(idx / ((long)Wo * Ho));
    float sy = (yo + 0.5f) * (float)Hi / (float)Ho - 0.5f;
    float sx = (xo + 0.5f) * (float)Wi / (float)Wo - 0.5f;
    int y0 = (int)floorf(sy); float fy = sy - y0;
    int x0 = (int)floorf(sx); float fx = sx - x0;
    int y1 = y0 + 1, x1 = x0 + 1;
    y0 = y0 < 0 ? 0 : (y0 > Hi - 1 ? Hi - 1 : y0);
    y1 = y1 < 0 ? 0 : (y1 > Hi - 1 ? Hi - 1 : y1);
    x0 = x0 < 0 ? 0 : (x0 > Wi - 1 ? Wi - 1 : x0);
    x1 = x1 < 0 ? 0 : (x1 > Wi - 1 ? Wi - 1 : x1);
    const float* ip = in + (size_t)p * Hi * Wi;
    float v = (1.f - fy) * ((1.f - fx) * ip[y0 * Wi + x0] + fx * ip[y0 * Wi + x1])
            +        fy  * ((1.f - fx) * ip[y1 * Wi + x0] + fx * ip[y1 * Wi + x1]);
    out[idx] = v;
}

// f3 [T][C][HW] * conf[T][HW] -> tokens f16 [(p*T+t)*C + c], conf_seq[p*T+t]
__global__ void k_tokens_build(const float* __restrict__ f3, const float* __restrict__ conf,
                               f16* __restrict__ tok, float* __restrict__ confseq,
                               int T, int C, int HW) {
    long idx = (long)blockIdx.x * blockDim.x + threadIdx.x;
    long total = (long)HW * T * C;
    if (idx >= total) return;
    int c = (int)(idx % C);
    int t = (int)((idx / C) % T);
    int p = (int)(idx / ((long)C * T));
    float cv = conf[(size_t)t * HW + p];
    tok[idx] = (f16)(f3[((size_t)t * C + c) * HW + p] * cv);
    if (c == 0) confseq[(size_t)p * T + t] = cv;
}

// g[nt,c] *= sigmoid(confseq[nt]*Ws[c] + bs[c])
__global__ void k_gate(float* __restrict__ g, const float* __restrict__ confseq,
                       const float* __restrict__ Ws, const float* __restrict__ bs,
                       long total, int C) {
    long i = (long)blockIdx.x * blockDim.x + threadIdx.x;
    if (i >= total) return;
    int c = (int)(i % C);
    long nt = i / C;
    g[i] *= sigmoid_f(confseq[nt] * Ws[c] + bs[c]);
}

// recurrent gate scan over T (tokens laid out [(p*T+t)*C + c])
__global__ void k_scan(const float* __restrict__ g, const float* __restrict__ xt,
                       float* __restrict__ ss, int Npix, int T, int C) {
    long i = (long)blockIdx.x * blockDim.x + threadIdx.x;
    if (i >= (long)Npix * C) return;
    int c = (int)(i % C);
    int p = (int)(i / C);
    float s = 0.f;
    for (int t = 0; t < T; ++t) {
        size_t o = ((size_t)p * T + t) * (size_t)C + c;
        float gt = g[o];
        s = (1.f - gt) * s + gt * xt[o];
        ss[o] = s;
    }
}

// mean over T -> y [C][HW] (f32 + f16)
__global__ void k_mean_tok(const float* __restrict__ tok, float* __restrict__ o32,
                           f16* __restrict__ o16, int Npix, int T, int C) {
    long i = (long)blockIdx.x * blockDim.x + threadIdx.x;
    if (i >= (long)Npix * C) return;
    int c = (int)(i % C);
    int p = (int)(i / C);
    float s = 0.f;
    for (int t = 0; t < T; ++t) s += tok[((size_t)p * T + t) * (size_t)C + c];
    s /= (float)T;
    o32[(size_t)c * Npix + p] = s;
    o16[(size_t)c * Npix + p] = (f16)s;
}

// confidence-weighted temporal average of f [T][C][HW] -> f16 [C][HW]
__global__ void k_skip_fuse(const float* __restrict__ f, const float* __restrict__ conf,
                            f16* __restrict__ out16, int T, int C, int HW) {
    long i = (long)blockIdx.x * blockDim.x + threadIdx.x;
    if (i >= (long)C * HW) return;
    int p = (int)(i % HW);
    int c = (int)(i / HW);
    float acc = 0.f, wsum = 0.f;
    for (int t = 0; t < T; ++t) {
        float cv = conf[(size_t)t * HW + p];
        acc  += f[((size_t)t * C + c) * HW + p] * cv;
        wsum += cv;
    }
    out16[i] = (f16)(acc / (wsum + 1e-6f));
}

// ---------------------------------------------------------------------------
// Host orchestration
// ---------------------------------------------------------------------------
static int host_numgroups(int c) {
    int ng = c / 8;
    if (ng < 1) ng = 1;
    if (ng > 32) ng = 32;
    while ((c % ng) != 0 && ng > 1) --ng;
    return ng;
}

struct CGA { const float *w, *g, *b; };
struct RES { CGA c1, c2; };
struct SSM { const float *Win, *Wout, *Wg, *bg, *Ws, *bs, *W1, *b1, *W2, *b2; };

extern "C" void kernel_launch(void* const* d_in, const int* in_sizes, int n_in,
                              void* d_out, int out_size, void* d_ws, size_t ws_size,
                              hipStream_t stream)
{
    (void)in_sizes; (void)n_in; (void)out_size;

    // ---- gather input pointers in setup_inputs() dict order ----
    int ii = 0;
    auto nxt = [&]() -> const float* { return (const float*)d_in[ii++]; };
    auto getCGA = [&]() { CGA c; c.w = nxt(); c.g = nxt(); c.b = nxt(); return c; };
    auto getRES = [&]() { RES r; r.c1 = getCGA(); r.c2 = getCGA(); return r; };
    auto getSSM = [&]() { SSM s; s.Win = nxt(); s.Wout = nxt(); s.Wg = nxt(); s.bg = nxt();
                          s.Ws = nxt(); s.bs = nxt(); s.W1 = nxt(); s.b1 = nxt();
                          s.W2 = nxt(); s.b2 = nxt(); return s; };

    const float* x = nxt();
    CGA stem_c = getCGA(); RES stem_r = getRES();
    CGA d1_c   = getCGA(); RES d1_r   = getRES();
    CGA d2_c   = getCGA(); RES d2_r   = getRES();
    CGA deep_c = getCGA(); RES deep_r = getRES();
    CGA cl1 = getCGA(), cl2 = getCGA(), cl3 = getCGA();
    const float* cl_wout = nxt(); const float* cl_bout = nxt();
    SSM ssm0 = getSSM(), ssm1 = getSSM();
    CGA post_c = getCGA(); RES post_r = getRES();
    CGA m4_c = getCGA(); RES m4_r = getRES();
    CGA u2_c = getCGA(); RES u2_r = getRES();
    CGA m2_c = getCGA(); RES m2_r = getRES();
    CGA u1_c = getCGA(); RES u1_r = getRES();
    CGA oh_c = getCGA();
    const float* dec_wout = nxt(); const float* dec_bout = nxt();

    // ---- two-ended workspace arena ----
    char* ws = (char*)d_ws;
    size_t lo = 0, hi = ws_size;
    auto alo = [&](size_t b) -> char* { char* p = ws + lo; lo += (b + 255) & ~(size_t)255; return p; };
    auto ahi = [&](size_t b) -> char* { size_t rb = (b + 255) & ~(size_t)255; hi -= rb; return ws + hi; };

    // ---- common launchers ----
    auto cvt = [&](const float* s, f16* dst, long n) {
        k_f32_to_f16<<<(unsigned)((n + 255) / 256), 256, 0, stream>>>(s, dst, n);
    };
    auto cvtN = [&](const float* s, long n) -> f16* {
        f16* p = (f16*)ahi((size_t)n * 2); cvt(s, p, n); return p;
    };
    float* tA = (float*)ahi((size_t)6 * 48 * 65536 * 4);   // biggest conv output
    auto conv = [&](const f16* in, const f16* w16, float* outp,
                    int Cin, int Hin, int Wi, int Cout, int ks, int st, int nimg) {
        int pad = ks / 2;
        int Ho = (Hin + 2 * pad - ks) / st + 1;
        int Wo = (Wi  + 2 * pad - ks) / st + 1;
        dim3 grid((unsigned)((Ho * Wo + 127) / 128), (unsigned)((Cout + 63) / 64), (unsigned)nimg);
        if (ks == 1)
            k_conv_gemm<1><<<grid, 256, 0, stream>>>(in, w16, outp, Cin, Hin, Wi, Cout, Ho, Wo, st);
        else
            k_conv_gemm<3><<<grid, 256, 0, stream>>>(in, w16, outp, Cin, Hin, Wi, Cout, Ho, Wo, st);
    };
    auto gn = [&](const float* xin, const float* g, const float* b, const float* resid,
                  float* o32, f16* o16, int C, int HWp, int nimg) {
        int ng = host_numgroups(C);
        dim3 grid((unsigned)ng, (unsigned)nimg);
        k_gn_gelu<<<grid, 256, 0, stream>>>(xin, g, b, resid, o32, o16, C, HWp, ng);
    };
    auto cga_run = [&](const f16* in16, const f16* w16, const float* g, const float* b,
                       const float* resid, float* o32, f16* o16,
                       int Cin, int Hin, int Wi, int Cout, int ks, int st, int nimg) {
        conv(in16, w16, tA, Cin, Hin, Wi, Cout, ks, st, nimg);
        int pad = ks / 2;
        int Ho = (Hin + 2 * pad - ks) / st + 1;
        int Wo = (Wi  + 2 * pad - ks) / st + 1;
        gn(tA, g, b, resid, o32, o16, Cout, Ho * Wo, nimg);
    };
    auto lin = [&](const f16* A, const f16* W, const float* bias, const float* resid,
                   float* outp, long M, int K, int N, int act) {
        dim3 grid((unsigned)((M + 127) / 128), (unsigned)((N + 63) / 64));
        k_lin_gemm<<<grid, 256, 0, stream>>>(A, W, bias, resid, outp, (int)M, K, N, act);
    };

    const int HWA = 65536, HWB = 16384, HWC = 4096;

    // ---- phase 0: f16 copies of input + all weights (persistent, top of arena) ----
    f16* xh = (f16*)ahi((size_t)6 * 3 * HWA * 2);
    cvt(x, xh, (long)6 * 3 * HWA);

    f16 *W_stem = cvtN(stem_c.w, 48 * 3 * 9);
    f16 *W_sr1 = cvtN(stem_r.c1.w, 48 * 48 * 9), *W_sr2 = cvtN(stem_r.c2.w, 48 * 48 * 9);
    f16 *W_d1  = cvtN(d1_c.w, 48 * 48 * 9);
    f16 *W_d1r1 = cvtN(d1_r.c1.w, 48 * 48 * 9), *W_d1r2 = cvtN(d1_r.c2.w, 48 * 48 * 9);
    f16 *W_d2  = cvtN(d2_c.w, 96 * 48 * 9);
    f16 *W_d2r1 = cvtN(d2_r.c1.w, 96 * 96 * 9), *W_d2r2 = cvtN(d2_r.c2.w, 96 * 96 * 9);
    f16 *W_dp  = cvtN(deep_c.w, 192 * 96 * 9);
    f16 *W_dpr1 = cvtN(deep_r.c1.w, 192 * 192 * 9), *W_dpr2 = cvtN(deep_r.c2.w, 192 * 192 * 9);
    f16 *W_cl1 = cvtN(cl1.w, 32 * 3 * 9), *W_cl2 = cvtN(cl2.w, 32 * 32 * 9), *W_cl3 = cvtN(cl3.w, 32 * 32 * 9);
    f16 *W_clo = cvtN(cl_wout, 32);
    f16 *W_post = cvtN(post_c.w, 192 * 192 * 9);
    f16 *W_pr1 = cvtN(post_r.c1.w, 192 * 192 * 9), *W_pr2 = cvtN(post_r.c2.w, 192 * 192 * 9);
    f16 *W_m4 = cvtN(m4_c.w, 192 * 288 * 9);
    f16 *W_m4r1 = cvtN(m4_r.c1.w, 192 * 192 * 9), *W_m4r2 = cvtN(m4_r.c2.w, 192 * 192 * 9);
    f16 *W_u2 = cvtN(u2_c.w, 96 * 192 * 9);
    f16 *W_u2r1 = cvtN(u2_r.c1.w, 96 * 96 * 9), *W_u2r2 = cvtN(u2_r.c2.w, 96 * 96 * 9);
    f16 *W_m2 = cvtN(m2_c.w, 96 * 144 * 9);
    f16 *W_m2r1 = cvtN(m2_r.c1.w, 96 * 96 * 9), *W_m2r2 = cvtN(m2_r.c2.w, 96 * 96 * 9);
    f16 *W_u1 = cvtN(u1_c.w, 48 * 96 * 9);
    f16 *W_u1r1 = cvtN(u1_r.c1.w, 48 * 48 * 9), *W_u1r2 = cvtN(u1_r.c2.w, 48 * 48 * 9);
    f16 *W_oh = cvtN(oh_c.w, 48 * 48 * 9);
    f16 *W_out = cvtN(dec_wout, 3 * 48 * 9);
    const SSM* sp[2] = { &ssm0, &ssm1 };
    f16 *S_win[2], *S_w1[2], *S_w2[2], *S_wg[2], *S_wo[2];
    for (int i = 0; i < 2; ++i) {
        S_win[i] = cvtN(sp[i]->Win, 192 * 192);
        S_w1[i]  = cvtN(sp[i]->W1, 384 * 192);
        S_w2[i]  = cvtN(sp[i]->W2, 192 * 384);
        S_wg[i]  = cvtN(sp[i]->Wg, 192 * 192);
        S_wo[i]  = cvtN(sp[i]->Wout, 192 * 192);
    }

    // ---- persistent activations ----
    float* f1f     = (float*)ahi((size_t)6 * 48 * HWB * 4);
    float* f2f     = (float*)ahi((size_t)6 * 96 * HWC * 4);
    float* f3f     = (float*)ahi((size_t)6 * 192 * HWC * 4);
    float* supf    = (float*)ahi((size_t)6 * HWC * 4);
    float* conf64  = (float*)ahi((size_t)6 * HWC * 4);
    float* conf128 = (float*)ahi((size_t)6 * HWB * 4);
    float* deepf   = (float*)ahi((size_t)192 * HWC * 4);

    const size_t m0 = lo;

    // ===================== encoder (6 frames stacked) =====================
    f16*   a0h = (f16*)alo((size_t)6 * 48 * HWA * 2);
    float* a0f = (float*)alo((size_t)6 * 48 * HWA * 4);
    cga_run(xh, W_stem, stem_c.g, stem_c.b, nullptr, a0f, a0h, 3, 256, 256, 48, 3, 1, 6);
    f16* r1h = (f16*)alo((size_t)6 * 48 * HWA * 2);
    cga_run(a0h, W_sr1, stem_r.c1.g, stem_r.c1.b, nullptr, nullptr, r1h, 48, 256, 256, 48, 3, 1, 6);
    f16* x0h = (f16*)alo((size_t)6 * 48 * HWA * 2);
    cga_run(r1h, W_sr2, stem_r.c2.g, stem_r.c2.b, a0f, nullptr, x0h, 48, 256, 256, 48, 3, 1, 6);
    // d1 (stride 2)
    conv(x0h, W_d1, tA, 48, 256, 256, 48, 3, 2, 6);
    lo = m0;
    f16*   a1h = (f16*)alo((size_t)6 * 48 * HWB * 2);
    float* a1f = (float*)alo((size_t)6 * 48 * HWB * 4);
    gn(tA, d1_c.g, d1_c.b, nullptr, a1f, a1h, 48, HWB, 6);
    f16* rb = (f16*)alo((size_t)6 * 48 * HWB * 2);
    cga_run(a1h, W_d1r1, d1_r.c1.g, d1_r.c1.b, nullptr, nullptr, rb, 48, 128, 128, 48, 3, 1, 6);
    f16* f1h = (f16*)alo((size_t)6 * 48 * HWB * 2);
    cga_run(rb, W_d1r2, d1_r.c2.g, d1_r.c2.b, a1f, f1f, f1h, 48, 128, 128, 48, 3, 1, 6);
    // d2 (stride 2)
    conv(f1h, W_d2, tA, 48, 128, 128, 96, 3, 2, 6);
    lo = m0;
    f16*   a2h = (f16*)alo((size_t)6 * 96 * HWC * 2);
    float* a2f = (float*)alo((size_t)6 * 96 * HWC * 4);
    gn(tA, d2_c.g, d2_c.b, nullptr, a2f, a2h, 96, HWC, 6);
    f16* rc = (f16*)alo((size_t)6 * 96 * HWC * 2);
    cga_run(a2h, W_d2r1, d2_r.c1.g, d2_r.c1.b, nullptr, nullptr, rc, 96, 64, 64, 96, 3, 1, 6);
    f16* f2h = (f16*)alo((size_t)6 * 96 * HWC * 2);
    cga_run(rc, W_d2r2, d2_r.c2.g, d2_r.c2.b, a2f, f2f, f2h, 96, 64, 64, 96, 3, 1, 6);
    // deep
    f16*   a3h = (f16*)alo((size_t)6 * 192 * HWC * 2);
    float* a3f = (float*)alo((size_t)6 * 192 * HWC * 4);
    cga_run(f2h, W_dp, deep_c.g, deep_c.b, nullptr, a3f, a3h, 96, 64, 64, 192, 3, 1, 6);
    f16* rd = (f16*)alo((size_t)6 * 192 * HWC * 2);
    cga_run(a3h, W_dpr1, deep_r.c1.g, deep_r.c1.b, nullptr, nullptr, rd, 192, 64, 64, 192, 3, 1, 6);
    cga_run(rd, W_dpr2, deep_r.c2.g, deep_r.c2.b, a3f, f3f, nullptr, 192, 64, 64, 192, 3, 1, 6);
    lo = m0;

    // ===================== cloud head =====================
    f16* ch1 = (f16*)alo((size_t)6 * 32 * HWA * 2);
    cga_run(xh, W_cl1, cl1.g, cl1.b, nullptr, nullptr, ch1, 3, 256, 256, 32, 3, 1, 6);
    f16* ch2 = (f16*)alo((size_t)6 * 32 * HWB * 2);
    cga_run(ch1, W_cl2, cl2.g, cl2.b, nullptr, nullptr, ch2, 32, 256, 256, 32, 3, 2, 6);
    f16* ch3 = (f16*)alo((size_t)6 * 32 * HWC * 2);
    cga_run(ch2, W_cl3, cl3.g, cl3.b, nullptr, nullptr, ch3, 32, 128, 128, 32, 3, 2, 6);
    conv(ch3, W_clo, supf, 32, 64, 64, 1, 1, 1, 6);
    k_bias_act<<<(unsigned)(((long)6 * HWC + 255) / 256), 256, 0, stream>>>(supf, cl_bout, 1, HWC, (long)6 * HWC, 2);
    lo = m0;

    // confidence maps
    k_conf<<<(unsigned)(((long)6 * HWC + 255) / 256), 256, 0, stream>>>(supf, conf64, (long)6 * HWC);
    k_resize<<<(unsigned)(((long)6 * HWB + 255) / 256), 256, 0, stream>>>(supf, conf128, 6, 64, 64, 128, 128);
    k_conf<<<(unsigned)(((long)6 * HWB + 255) / 256), 256, 0, stream>>>(conf128, conf128, (long)6 * HWB);

    // ===================== temporal fuse (SSM) =====================
    const long NT = (long)HWC * 6;     // 24576 tokens
    f16*   tokh    = (f16*)alo((size_t)NT * 192 * 2);
    float* confseq = (float*)alo((size_t)NT * 4);
    k_tokens_build<<<(unsigned)(((long)HWC * 6 * 192 + 255) / 256), 256, 0, stream>>>(
        f3f, conf64, tokh, confseq, 6, 192, HWC);
    float* x32  = (float*)alo((size_t)NT * 192 * 4);  f16* x16  = (f16*)alo((size_t)NT * 192 * 2);
    float* h32  = (float*)alo((size_t)NT * 384 * 4);  f16* h16  = (f16*)alo((size_t)NT * 384 * 2);
    float* xt32 = (float*)alo((size_t)NT * 192 * 4);  f16* xt16 = (f16*)alo((size_t)NT * 192 * 2);
    float* g32  = (float*)alo((size_t)NT * 192 * 4);
    float* ss32 = (float*)alo((size_t)NT * 192 * 4);  f16* ss16 = (f16*)alo((size_t)NT * 192 * 2);
    float* nt32 = (float*)alo((size_t)NT * 192 * 4);
    for (int bk = 0; bk < 2; ++bk) {
        lin(tokh, S_win[bk], nullptr, nullptr, x32, NT, 192, 192, 0);
        cvt(x32, x16, NT * 192);
        lin(x16, S_w1[bk], sp[bk]->b1, nullptr, h32, NT, 192, 384, 1);
        cvt(h32, h16, NT * 384);
        lin(h16, S_w2[bk], sp[bk]->b2, x32, xt32, NT, 384, 192, 0);
        cvt(xt32, xt16, NT * 192);
        lin(xt16, S_wg[bk], sp[bk]->bg, nullptr, g32, NT, 192, 192, 2);
        k_gate<<<(unsigned)((NT * 192 + 255) / 256), 256, 0, stream>>>(g32, confseq, sp[bk]->Ws, sp[bk]->bs, NT * 192, 192);
        k_scan<<<(unsigned)(((long)HWC * 192 + 255) / 256), 256, 0, stream>>>(g32, xt32, ss32, HWC, 6, 192);
        cvt(ss32, ss16, NT * 192);
        lin(ss16, S_wo[bk], nullptr, nullptr, nt32, NT, 192, 192, 0);
        cvt(nt32, tokh, NT * 192);
    }
    float* y32 = (float*)alo((size_t)192 * HWC * 4);
    f16*   y16 = (f16*)alo((size_t)192 * HWC * 2);
    k_mean_tok<<<(unsigned)(((long)HWC * 192 + 255) / 256), 256, 0, stream>>>(nt32, y32, y16, HWC, 6, 192);
    // post conv + res
    f16*   pch = (f16*)alo((size_t)192 * HWC * 2);
    float* pcf = (float*)alo((size_t)192 * HWC * 4);
    cga_run(y16, W_post, post_c.g, post_c.b, nullptr, pcf, pch, 192, 64, 64, 192, 3, 1, 1);
    f16* prh = (f16*)alo((size_t)192 * HWC * 2);
    cga_run(pch, W_pr1, post_r.c1.g, post_r.c1.b, nullptr, nullptr, prh, 192, 64, 64, 192, 3, 1, 1);
    cga_run(prh, W_pr2, post_r.c2.g, post_r.c2.b, pcf, deepf, nullptr, 192, 64, 64, 192, 3, 1, 1);
    lo = m0;

    // ===================== decoder =====================
    f16* cat4 = (f16*)alo((size_t)288 * HWC * 2);
    cvt(deepf, cat4, (long)192 * HWC);
    k_skip_fuse<<<(unsigned)(((long)96 * HWC + 255) / 256), 256, 0, stream>>>(
        f2f, conf64, cat4 + (size_t)192 * HWC, 6, 96, HWC);
    f16*   m4h = (f16*)alo((size_t)192 * HWC * 2);
    float* m4f = (float*)alo((size_t)192 * HWC * 4);
    cga_run(cat4, W_m4, m4_c.g, m4_c.b, nullptr, m4f, m4h, 288, 64, 64, 192, 3, 1, 1);
    f16* mr = (f16*)alo((size_t)192 * HWC * 2);
    cga_run(m4h, W_m4r1, m4_r.c1.g, m4_r.c1.b, nullptr, nullptr, mr, 192, 64, 64, 192, 3, 1, 1);
    float* d4f = (float*)alo((size_t)192 * HWC * 4);
    cga_run(mr, W_m4r2, m4_r.c2.g, m4_r.c2.b, m4f, d4f, nullptr, 192, 64, 64, 192, 3, 1, 1);
    // upsample 64 -> 128
    float* u128f = (float*)alo((size_t)192 * HWB * 4);
    k_resize<<<(unsigned)(((long)192 * HWB + 255) / 256), 256, 0, stream>>>(d4f, u128f, 192, 64, 64, 128, 128);
    f16* u128h = (f16*)alo((size_t)192 * HWB * 2);
    cvt(u128f, u128h, (long)192 * HWB);
    f16*   u2h = (f16*)alo((size_t)96 * HWB * 2);
    float* u2f = (float*)alo((size_t)96 * HWB * 4);
    cga_run(u128h, W_u2, u2_c.g, u2_c.b, nullptr, u2f, u2h, 192, 128, 128, 96, 3, 1, 1);
    f16* ur = (f16*)alo((size_t)96 * HWB * 2);
    cga_run(u2h, W_u2r1, u2_r.c1.g, u2_r.c1.b, nullptr, nullptr, ur, 96, 128, 128, 96, 3, 1, 1);
    f16* cat2 = (f16*)alo((size_t)144 * HWB * 2);
    cga_run(ur, W_u2r2, u2_r.c2.g, u2_r.c2.b, u2f, nullptr, cat2, 96, 128, 128, 96, 3, 1, 1);
    k_skip_fuse<<<(unsigned)(((long)48 * HWB + 255) / 256), 256, 0, stream>>>(
        f1f, conf128, cat2 + (size_t)96 * HWB, 6, 48, HWB);
    f16*   m2h = (f16*)alo((size_t)96 * HWB * 2);
    float* m2f_ = (float*)alo((size_t)96 * HWB * 4);
    cga_run(cat2, W_m2, m2_c.g, m2_c.b, nullptr, m2f_, m2h, 144, 128, 128, 96, 3, 1, 1);
    f16* mr2 = (f16*)alo((size_t)96 * HWB * 2);
    cga_run(m2h, W_m2r1, m2_r.c1.g, m2_r.c1.b, nullptr, nullptr, mr2, 96, 128, 128, 96, 3, 1, 1);
    float* m2o = (float*)alo((size_t)96 * HWB * 4);
    cga_run(mr2, W_m2r2, m2_r.c2.g, m2_r.c2.b, m2f_, m2o, nullptr, 96, 128, 128, 96, 3, 1, 1);
    // upsample 128 -> 256
    float* r256f = (float*)alo((size_t)96 * HWA * 4);
    k_resize<<<(unsigned)(((long)96 * HWA + 255) / 256), 256, 0, stream>>>(m2o, r256f, 96, 128, 128, 256, 256);
    f16* r256h = (f16*)alo((size_t)96 * HWA * 2);
    cvt(r256f, r256h, (long)96 * HWA);
    f16*   u1h = (f16*)alo((size_t)48 * HWA * 2);
    float* u1f = (float*)alo((size_t)48 * HWA * 4);
    cga_run(r256h, W_u1, u1_c.g, u1_c.b, nullptr, u1f, u1h, 96, 256, 256, 48, 3, 1, 1);
    f16* ur1 = (f16*)alo((size_t)48 * HWA * 2);
    cga_run(u1h, W_u1r1, u1_r.c1.g, u1_r.c1.b, nullptr, nullptr, ur1, 48, 256, 256, 48, 3, 1, 1);
    f16* u1o = (f16*)alo((size_t)48 * HWA * 2);
    cga_run(ur1, W_u1r2, u1_r.c2.g, u1_r.c2.b, u1f, nullptr, u1o, 48, 256, 256, 48, 3, 1, 1);
    f16* ohh = (f16*)alo((size_t)48 * HWA * 2);
    cga_run(u1o, W_oh, oh_c.g, oh_c.b, nullptr, nullptr, ohh, 48, 256, 256, 48, 3, 1, 1);
    // final 3x3 conv straight into d_out, then bias
    conv(ohh, W_out, (float*)d_out, 48, 256, 256, 3, 3, 1, 1);
    k_bias_act<<<(unsigned)(((long)3 * HWA + 255) / 256), 256, 0, stream>>>(
        (float*)d_out, dec_bout, 3, HWA, (long)3 * HWA, 0);
}
